// ImprovedGATLayer_9423158247920
// MI455X (gfx1250) — compile-verified, matrix operand order checked by reading, not run on hardware
//
#include <hip/hip_runtime.h>
#include <hip/hip_bf16.h>

// Problem constants (match reference)
#define N_NODES 20000
#define N_EDGES 320000
#define C_DIM   128
#define H_HEADS 5
#define HC      (H_HEADS * C_DIM)      // 640
#define E_TOT   (N_EDGES + N_NODES)    // 340000 (with self loops)
#define NEG_SLOPE 0.2f
#define EPS_F 1e-16f

typedef __attribute__((ext_vector_type(16))) _Float16 v16h;
typedef __attribute__((ext_vector_type(8)))  float    v8f;

// Monotone order-preserving float<->uint encoding for atomicMax-based segment max
__device__ __forceinline__ unsigned fenc(float f) {
    unsigned u = __float_as_uint(f);
    return (u & 0x80000000u) ? ~u : (u | 0x80000000u);
}
__device__ __forceinline__ float fdec(unsigned u) {
    unsigned v = (u & 0x80000000u) ? (u & 0x7FFFFFFFu) : ~u;
    return __uint_as_float(v);
}

__device__ __forceinline__ float leaky(float x) {
    return x > 0.0f ? x : NEG_SLOPE * x;
}

// ---------------------------------------------------------------------------
// Kernel 0: zero-init accumulators (d_out is the scatter accumulator)
// ---------------------------------------------------------------------------
__global__ void gat_init(float* __restrict__ out, float* __restrict__ denom,
                         unsigned* __restrict__ emax) {
    int i = blockIdx.x * blockDim.x + threadIdx.x;
    if (i < N_NODES * C_DIM) out[i] = 0.0f;
    if (i < N_NODES * H_HEADS) { denom[i] = 0.0f; emax[i] = 0u; /* enc(-inf) floor */ }
}

// ---------------------------------------------------------------------------
// Kernel 1: h = X @ W  via WMMA f32_16x16x32_f16 with error-compensated
// f16 split (ahi*bhi + ahi*blo + alo*bhi ~ fp32 accuracy).
//
// The f32->f16 hi/lo split happens ONCE at staging time; the halves are
// written to LDS pre-swizzled into the exact CDNA5 wave32 fragment layout
// (lanes 0-15 hold K {0..7,16..23}, lanes 16-31 hold K {8..15,24..31}),
// so the hot loop is pure ds_load_b128 + v_wmma with no conversion VALU.
//
// Block = 128 threads = 4 waves; each wave owns one 16x16 output tile,
// block covers 16 rows x 64 cols.  LDS: 40 KB.
// ---------------------------------------------------------------------------
__global__ __launch_bounds__(128) void gat_gemm_wmma(
    const float* __restrict__ X,    // [N, 128] row-major
    const float* __restrict__ W,    // [128, 640] row-major
    float* __restrict__ Hout)       // [N, 640] row-major
{
    // Fragment-major LDS: A[kstep][lane][16], B[kstep][col][grp*16+j]
    __shared__ __align__(32) _Float16 Ahi[4 * 32 * 16];   // 4 KB
    __shared__ __align__(32) _Float16 Alo[4 * 32 * 16];   // 4 KB
    __shared__ __align__(32) _Float16 Bhi[4 * 64 * 32];   // 16 KB
    __shared__ __align__(32) _Float16 Blo[4 * 64 * 32];   // 16 KB

    const int tid    = threadIdx.x;
    const int wave   = tid >> 5;
    const int lane   = tid & 31;
    const int tile_m = blockIdx.x * 16;
    const int nbase  = blockIdx.y * 64;

    // Stage A tile (16 rows x 128 K), coalesced global reads, split + swizzle
    #pragma unroll
    for (int idx = tid; idx < 16 * 128; idx += 128) {
        int m = idx >> 7, k = idx & 127;
        float a = X[(tile_m + m) * C_DIM + k];
        _Float16 hi = (_Float16)a;
        _Float16 lo = (_Float16)(a - (float)hi);
        int s   = k >> 5, kk = k & 31;
        int grp = (kk >> 3) & 1;
        int j   = (kk & 7) + ((kk & 16) ? 8 : 0);
        int off = ((s * 32) + (grp * 16 + m)) * 16 + j;
        Ahi[off] = hi; Alo[off] = lo;
    }
    // Stage B tile (128 K x 64 cols), coalesced global reads, split + swizzle
    for (int idx = tid; idx < 128 * 64; idx += 128) {
        int k = idx >> 6, c = idx & 63;
        float b = W[k * HC + nbase + c];
        _Float16 hi = (_Float16)b;
        _Float16 lo = (_Float16)(b - (float)hi);
        int s   = k >> 5, kk = k & 31;
        int grp = (kk >> 3) & 1;
        int j   = (kk & 7) + ((kk & 16) ? 8 : 0);
        int off = ((s * 64) + c) * 32 + grp * 16 + j;
        Bhi[off] = hi; Blo[off] = lo;
    }
    __syncthreads();

    const int n    = lane & 15;
    const int grp  = lane >> 4;
    const int ncol = (wave << 4) + n;     // B column within 64-wide tile

    v8f acc = {};

    #pragma unroll
    for (int s = 0; s < 4; ++s) {         // K = 128 in 4 steps of 32
        const v16h ahi = *(const v16h*)&Ahi[(s * 32 + lane) * 16];
        const v16h alo = *(const v16h*)&Alo[(s * 32 + lane) * 16];
        const v16h bhi = *(const v16h*)&Bhi[(s * 64 + ncol) * 32 + grp * 16];
        const v16h blo = *(const v16h*)&Blo[(s * 64 + ncol) * 32 + grp * 16];
        acc = __builtin_amdgcn_wmma_f32_16x16x32_f16(false, ahi, false, bhi, (short)0, acc, false, false);
        acc = __builtin_amdgcn_wmma_f32_16x16x32_f16(false, ahi, false, blo, (short)0, acc, false, false);
        acc = __builtin_amdgcn_wmma_f32_16x16x32_f16(false, alo, false, bhi, (short)0, acc, false, false);
    }

    // D layout: VGPR r -> row r (lanes 0-15) / row r+8 (lanes 16-31), col = lane%16
    const int col = nbase + (wave << 4) + n;
    #pragma unroll
    for (int r = 0; r < 8; ++r) {
        int row = tile_m + r + (grp ? 8 : 0);
        Hout[row * HC + col] = acc[r];
    }
}

// ---------------------------------------------------------------------------
// Kernel 2: alpha_src/alpha_dst = <h[n,h,:], att_src/dst[h,:]>  (wave per pair)
// ---------------------------------------------------------------------------
__global__ __launch_bounds__(256) void gat_alpha(
    const float* __restrict__ Hfeat, const float* __restrict__ att_src,
    const float* __restrict__ att_dst, float* __restrict__ a_src,
    float* __restrict__ a_dst)
{
    int wg   = (blockIdx.x * blockDim.x + threadIdx.x) >> 5;
    int lane = threadIdx.x & 31;
    if (wg >= N_NODES * H_HEADS) return;
    int n  = wg / H_HEADS;
    int hd = wg % H_HEADS;
    const float4* hv = reinterpret_cast<const float4*>(Hfeat + (size_t)n * HC + hd * C_DIM);
    const float4* as = reinterpret_cast<const float4*>(att_src + hd * C_DIM);
    const float4* ad = reinterpret_cast<const float4*>(att_dst + hd * C_DIM);
    float4 v = hv[lane], s = as[lane], d = ad[lane];
    float ss = v.x * s.x + v.y * s.y + v.z * s.z + v.w * s.w;
    float dd = v.x * d.x + v.y * d.y + v.z * d.z + v.w * d.w;
    #pragma unroll
    for (int off = 16; off > 0; off >>= 1) {
        ss += __shfl_down(ss, off, 32);
        dd += __shfl_down(dd, off, 32);
    }
    if (lane == 0) { a_src[wg] = ss; a_dst[wg] = dd; }
}

// ---------------------------------------------------------------------------
// Kernel 3: per-edge segment max (ordered-uint atomicMax)
// ---------------------------------------------------------------------------
__global__ void gat_edge_max(const int* __restrict__ ei,
                             const float* __restrict__ a_src,
                             const float* __restrict__ a_dst,
                             unsigned* __restrict__ emax)
{
    int e = blockIdx.x * blockDim.x + threadIdx.x;
    if (e >= E_TOT) return;
    int s, d;
    if (e < N_EDGES) { s = ei[e]; d = ei[N_EDGES + e]; } else { s = d = e - N_EDGES; }
    #pragma unroll
    for (int h = 0; h < H_HEADS; ++h) {
        float x = leaky(a_src[s * H_HEADS + h] + a_dst[d * H_HEADS + h]);
        atomicMax(&emax[d * H_HEADS + h], fenc(x));
    }
}

// ---------------------------------------------------------------------------
// Kernel 4: per-edge exp + segment sum (denominator)
// ---------------------------------------------------------------------------
__global__ void gat_edge_sum(const int* __restrict__ ei,
                             const float* __restrict__ a_src,
                             const float* __restrict__ a_dst,
                             const unsigned* __restrict__ emax,
                             float* __restrict__ denom)
{
    int e = blockIdx.x * blockDim.x + threadIdx.x;
    if (e >= E_TOT) return;
    int s, d;
    if (e < N_EDGES) { s = ei[e]; d = ei[N_EDGES + e]; } else { s = d = e - N_EDGES; }
    #pragma unroll
    for (int h = 0; h < H_HEADS; ++h) {
        float x  = leaky(a_src[s * H_HEADS + h] + a_dst[d * H_HEADS + h]);
        float ex = __expf(x - fdec(emax[d * H_HEADS + h]));
        atomicAdd(&denom[d * H_HEADS + h], ex);
    }
}

// ---------------------------------------------------------------------------
// Kernel 5: weighted scatter with fused head-mean.  One wave32 per edge;
// lanes 0..4 compute the 5 head weights, broadcast via shfl, every lane
// handles 4 channels -> 128 atomic f32 adds per edge (h is L2-resident).
// ---------------------------------------------------------------------------
__global__ __launch_bounds__(256) void gat_scatter(
    const int* __restrict__ ei, const float* __restrict__ Hfeat,
    const float* __restrict__ a_src, const float* __restrict__ a_dst,
    const unsigned* __restrict__ emax, const float* __restrict__ denom,
    float* __restrict__ out)
{
    int wg   = (blockIdx.x * blockDim.x + threadIdx.x) >> 5;
    int lane = threadIdx.x & 31;
    if (wg >= E_TOT) return;
    int s, d;
    if (wg < N_EDGES) { s = ei[wg]; d = ei[N_EDGES + wg]; } else { s = d = wg - N_EDGES; }

    float w = 0.0f;
    if (lane < H_HEADS) {
        int h    = lane;
        float x  = leaky(a_src[s * H_HEADS + h] + a_dst[d * H_HEADS + h]);
        float ex = __expf(x - fdec(emax[d * H_HEADS + h]));
        w = ex / (denom[d * H_HEADS + h] + EPS_F) * (1.0f / H_HEADS);
    }
    float w0 = __shfl(w, 0, 32), w1 = __shfl(w, 1, 32), w2 = __shfl(w, 2, 32);
    float w3 = __shfl(w, 3, 32), w4 = __shfl(w, 4, 32);

    const float4* hv = reinterpret_cast<const float4*>(Hfeat);
    size_t base = (size_t)s * (HC / 4);           // s*160 float4's
    float4 v0 = hv[base + 0 * 32 + lane];
    float4 v1 = hv[base + 1 * 32 + lane];
    float4 v2 = hv[base + 2 * 32 + lane];
    float4 v3 = hv[base + 3 * 32 + lane];
    float4 v4 = hv[base + 4 * 32 + lane];

    float ax = w0 * v0.x + w1 * v1.x + w2 * v2.x + w3 * v3.x + w4 * v4.x;
    float ay = w0 * v0.y + w1 * v1.y + w2 * v2.y + w3 * v3.y + w4 * v4.y;
    float az = w0 * v0.z + w1 * v1.z + w2 * v2.z + w3 * v3.z + w4 * v4.z;
    float aw = w0 * v0.w + w1 * v1.w + w2 * v2.w + w3 * v3.w + w4 * v4.w;

    float* o = out + (size_t)d * C_DIM + lane * 4;
    atomicAdd(o + 0, ax);
    atomicAdd(o + 1, ay);
    atomicAdd(o + 2, az);
    atomicAdd(o + 3, aw);
}

// ---------------------------------------------------------------------------
// Kernel 6: residual + bias (in-place on the accumulator = d_out)
// ---------------------------------------------------------------------------
__global__ void gat_finalize(const float* __restrict__ X,
                             const float* __restrict__ bias,
                             float* __restrict__ out)
{
    int i = blockIdx.x * blockDim.x + threadIdx.x;
    if (i >= N_NODES * C_DIM) return;
    out[i] = X[i] + out[i] + bias[i & (C_DIM - 1)];
}

// ---------------------------------------------------------------------------
extern "C" void kernel_launch(void* const* d_in, const int* in_sizes, int n_in,
                              void* d_out, int out_size, void* d_ws, size_t ws_size,
                              hipStream_t stream) {
    const float* X       = (const float*)d_in[0];  // [N,128]
    const float* W       = (const float*)d_in[1];  // [128,640]
    const float* att_src = (const float*)d_in[2];  // [5,128]
    const float* att_dst = (const float*)d_in[3];  // [5,128]
    const float* bias    = (const float*)d_in[4];  // [128]
    const int*   ei      = (const int*)d_in[5];    // [2,E] flattened
    float* out = (float*)d_out;

    // Workspace layout (floats): h | a_src | a_dst | emax(uint) | denom  (~53 MB)
    float*    h     = (float*)d_ws;
    float*    a_src = h + (size_t)N_NODES * HC;
    float*    a_dst = a_src + (size_t)N_NODES * H_HEADS;
    unsigned* emax  = (unsigned*)(a_dst + (size_t)N_NODES * H_HEADS);
    float*    denom = (float*)(emax + (size_t)N_NODES * H_HEADS);

    const int NC = N_NODES * C_DIM;

    gat_init<<<(NC + 255) / 256, 256, 0, stream>>>(out, denom, emax);

    dim3 ggrid(N_NODES / 16, HC / 64);   // 1250 x 10
    gat_gemm_wmma<<<ggrid, 128, 0, stream>>>(X, W, h);

    gat_alpha<<<(N_NODES * H_HEADS) / 8, 256, 0, stream>>>(h, att_src, att_dst, a_src, a_dst);

    gat_edge_max<<<(E_TOT + 255) / 256, 256, 0, stream>>>(ei, a_src, a_dst, emax);
    gat_edge_sum<<<(E_TOT + 255) / 256, 256, 0, stream>>>(ei, a_src, a_dst, emax, denom);

    gat_scatter<<<E_TOT / 8, 256, 0, stream>>>(ei, h, a_src, a_dst, emax, denom, out);

    gat_finalize<<<(NC + 255) / 256, 256, 0, stream>>>(X, bias, out);
}
